// GraphModel_13194139533473
// MI455X (gfx1250) — matrix operand order repaired and checked by reading, not compiled, not run on hardware
//
#include <hip/hip_runtime.h>
#include <hip/hip_bf16.h>

// ---------------- types ----------------
typedef __attribute__((ext_vector_type(16))) __bf16 v16bf;
typedef __attribute__((ext_vector_type(8)))  __bf16 v8bf;
typedef __attribute__((ext_vector_type(8)))  float  v8f;

#define NCLS 12
#define NN   144        // NCLS*NCLS
#define C    256
#define B    1024
#define NODE_ROWS (B * NCLS)      // 12288
#define EDGE_ROWS (B * NN)        // 147456
#define EPS 1e-5f

__device__ __forceinline__ unsigned short f2bfu(float f) {
    unsigned u = __builtin_bit_cast(unsigned, f);
    unsigned r = (u + 0x7FFFu + ((u >> 16) & 1u)) >> 16;   // RTNE
    return (unsigned short)r;
}
__device__ __forceinline__ __bf16 f2bf(float f) {
    unsigned short h = f2bfu(f);
    return __builtin_bit_cast(__bf16, h);
}
__device__ __forceinline__ float bf2f(__bf16 b) {
    unsigned short h = __builtin_bit_cast(unsigned short, b);
    unsigned u = ((unsigned)h) << 16;
    return __builtin_bit_cast(float, u);
}

// 16x32 bf16 A-fragment: lane = half*16+m,
// elems 0..15 = K[k0+16h .. +15]
__device__ __forceinline__ v16bf load_a_bf(const __bf16* __restrict__ arow, int k0, int half) {
    return *(const v16bf*)(arow + k0 + 16 * half);
}

// ---------------- fp32 -> bf16 bulk convert (4 elems/thread) ----------------
__global__ void convert_bf16_kernel(const float* __restrict__ in, __bf16* __restrict__ out,
                                    unsigned n4) {
    unsigned i = blockIdx.x * blockDim.x + threadIdx.x;
    if (i < n4) {
        float4 f = ((const float4*)in)[i];
        uint2 p;
        p.x = (unsigned)f2bfu(f.x) | ((unsigned)f2bfu(f.y) << 16);
        p.y = (unsigned)f2bfu(f.z) | ((unsigned)f2bfu(f.w) << 16);
        ((uint2*)out)[i] = p;
    }
}

// ---------------- node GEMM: 2 row-tiles x 4 weights, shared A fragments ----------------
// grid(NODE_ROWS/32, 2) block(32,8): wave handles 32x16 output for 4 weights.
__global__ __launch_bounds__(256) void gemm_node4_kernel(
    const __bf16* __restrict__ Xbf,
    const __bf16* __restrict__ WA, const __bf16* __restrict__ WB,
    const __bf16* __restrict__ WV, const __bf16* __restrict__ WU,
    float* __restrict__ OA, float* __restrict__ OB,
    float* __restrict__ OV, float* __restrict__ OU)
{
    const int lane = threadIdx.x;
    const int half = lane >> 4;
    const int m16  = lane & 15;
    const int rowBase = blockIdx.x * 32;
    const int colBase = (blockIdx.y * 8 + threadIdx.y) * 16;

    const __bf16* arow0 = Xbf + (size_t)(rowBase + m16) * C;
    const __bf16* arow1 = arow0 + (size_t)16 * C;
    const size_t brow = (size_t)(colBase + m16) * C;

    v8f accA0 = {}; v8f accA1 = {};
    v8f accB0 = {}; v8f accB1 = {};
    v8f accV0 = {}; v8f accV1 = {};
    v8f accU0 = {}; v8f accU1 = {};

#pragma unroll
    for (int k0 = 0; k0 < C; k0 += 32) {
        v16bf a0 = load_a_bf(arow0, k0, half);
        v16bf a1 = load_a_bf(arow1, k0, half);
        const int boff = k0 + 16 * half;
        v16bf bA = *(const v16bf*)(WA + brow + boff);
        accA0 = __builtin_amdgcn_wmma_f32_16x16x32_bf16(false, a0, false, bA, (short)0, accA0, false, false);
        accA1 = __builtin_amdgcn_wmma_f32_16x16x32_bf16(false, a1, false, bA, (short)0, accA1, false, false);
        v16bf bB = *(const v16bf*)(WB + brow + boff);
        accB0 = __builtin_amdgcn_wmma_f32_16x16x32_bf16(false, a0, false, bB, (short)0, accB0, false, false);
        accB1 = __builtin_amdgcn_wmma_f32_16x16x32_bf16(false, a1, false, bB, (short)0, accB1, false, false);
        v16bf bV = *(const v16bf*)(WV + brow + boff);
        accV0 = __builtin_amdgcn_wmma_f32_16x16x32_bf16(false, a0, false, bV, (short)0, accV0, false, false);
        accV1 = __builtin_amdgcn_wmma_f32_16x16x32_bf16(false, a1, false, bV, (short)0, accV1, false, false);
        v16bf bU = *(const v16bf*)(WU + brow + boff);
        accU0 = __builtin_amdgcn_wmma_f32_16x16x32_bf16(false, a0, false, bU, (short)0, accU0, false, false);
        accU1 = __builtin_amdgcn_wmma_f32_16x16x32_bf16(false, a1, false, bU, (short)0, accU1, false, false);
    }

#pragma unroll
    for (int v = 0; v < 8; ++v) {
        size_t o0 = (size_t)(rowBase + v + 8 * half) * C + colBase + m16;
        size_t o1 = o0 + (size_t)16 * C;
        OA[o0] = accA0[v]; OA[o1] = accA1[v];
        OB[o0] = accB0[v]; OB[o1] = accB1[v];
        OV[o0] = accV0[v]; OV[o1] = accV1[v];
        OU[o0] = accU0[v]; OU[o1] = accU1[v];
    }
}

// ---------------- edge GEMM: 4 row-tiles share one B fragment ----------------
// m = edge@WE^T + Vix[b,i] + Vjx[b,j] -> bf16.  grid(EDGE_ROWS/64, 2) block(32,8)
__global__ __launch_bounds__(256) void gemm_edge_kernel(
    const __bf16* __restrict__ Ebf, const __bf16* __restrict__ WE,
    const float* __restrict__ Vix, const float* __restrict__ Vjx,
    __bf16* __restrict__ Mbf)
{
    const int lane = threadIdx.x;
    const int half = lane >> 4;
    const int m16  = lane & 15;
    const int rowBase = blockIdx.x * 64;
    const int colBase = (blockIdx.y * 8 + threadIdx.y) * 16;

    const __bf16* arow[4];
#pragma unroll
    for (int r = 0; r < 4; ++r)
        arow[r] = Ebf + (size_t)(rowBase + 16 * r + m16) * C;
    const size_t brow = (size_t)(colBase + m16) * C;

    v8f acc[4] = {{}, {}, {}, {}};

#pragma unroll
    for (int k0 = 0; k0 < C; k0 += 32) {
        v16bf bf = *(const v16bf*)(WE + brow + k0 + 16 * half);
#pragma unroll
        for (int r = 0; r < 4; ++r) {
            v16bf af = load_a_bf(arow[r], k0, half);
            acc[r] = __builtin_amdgcn_wmma_f32_16x16x32_bf16(false, af, false, bf, (short)0, acc[r], false, false);
        }
    }

    const int col = colBase + m16;
#pragma unroll
    for (int r = 0; r < 4; ++r) {
#pragma unroll
        for (int v = 0; v < 8; ++v) {
            int row = rowBase + 16 * r + v + 8 * half;   // = b*144 + i*12 + j
            int b  = row / NN;
            int rr = row - b * NN;
            int i  = rr / NCLS;
            int j  = rr - i * NCLS;
            float val = acc[r][v]
                      + Vix[((size_t)b * NCLS + i) * C + col]
                      + Vjx[((size_t)b * NCLS + j) * C + col];
            Mbf[(size_t)row * C + col] = f2bf(val);
        }
    }
}

// ---------------- BN stats over (batch, C) per edge-feature e ----------------
__global__ void stats_e_kernel(const __bf16* __restrict__ Mbf,
                               float* __restrict__ muE, float* __restrict__ rsE) {
    const int e = blockIdx.x, t = threadIdx.x;
    float s = 0.f, q = 0.f;
    const __bf16* base = Mbf + (size_t)e * C + t;
    for (int b = 0; b < B; ++b) {
        float v = bf2f(base[(size_t)b * NN * C]);
        s += v; q += v * v;
    }
    __shared__ float sS[256], sQ[256];
    sS[t] = s; sQ[t] = q; __syncthreads();
    for (int o = 128; o > 0; o >>= 1) {
        if (t < o) { sS[t] += sS[t + o]; sQ[t] += sQ[t + o]; }
        __syncthreads();
    }
    if (t == 0) {
        const float inv = 1.f / (float)(B * C);
        float mu = sS[0] * inv;
        float var = sQ[0] * inv - mu * mu;
        muE[e] = mu; rsE[e] = rsqrtf(var + EPS);
    }
}

__global__ void stats_v_kernel(const float* __restrict__ Ux, const float* __restrict__ Agg,
                               float* __restrict__ muV, float* __restrict__ rsV) {
    const int i = blockIdx.x, t = threadIdx.x;
    float s = 0.f, q = 0.f;
    const size_t stride = (size_t)NCLS * C;
    size_t off = (size_t)i * C + t;
    for (int b = 0; b < B; ++b, off += stride) {
        float v = Ux[off] + Agg[off];
        s += v; q += v * v;
    }
    __shared__ float sS[256], sQ[256];
    sS[t] = s; sQ[t] = q; __syncthreads();
    for (int o = 128; o > 0; o >>= 1) {
        if (t < o) { sS[t] += sS[t + o]; sQ[t] += sQ[t + o]; }
        __syncthreads();
    }
    if (t == 0) {
        const float inv = 1.f / (float)(B * C);
        float mu = sS[0] * inv;
        float var = sQ[0] * inv - mu * mu;
        muV[i] = mu; rsV[i] = rsqrtf(var + EPS);
    }
}

// ---------------- edge residual update: Eout = Ein + relu(bn(m)), fused bf16 copy ----------------
__global__ void edge_update_kernel(const float* __restrict__ Ein, const __bf16* __restrict__ Mbf,
                                   const float* __restrict__ Ge, const float* __restrict__ Be,
                                   const float* __restrict__ MuE, const float* __restrict__ RsE,
                                   float* __restrict__ Eout, __bf16* __restrict__ Ebf) {
    unsigned idx = blockIdx.x * 256u + threadIdx.x;    // 0 .. EDGE_ROWS*C-1
    unsigned row = idx >> 8;
    unsigned e = row % NN;
    float m = bf2f(Mbf[idx]);
    float bn = Ge[e] * (m - MuE[e]) * RsE[e] + Be[e];
    float v = Ein[idx] + fmaxf(bn, 0.f);
    Eout[idx] = v;
    Ebf[idx]  = f2bf(v);
}

// ---------------- softmax over j + aggregation with Ujx ----------------
__global__ void agg_kernel(const float* __restrict__ E, const float* __restrict__ Ujx,
                           float* __restrict__ Agg) {
    unsigned idx = blockIdx.x * 256u + threadIdx.x;    // 0 .. NODE_ROWS*C-1
    unsigned c = idx & 255u;
    unsigned bi = idx >> 8;
    unsigned b = bi / NCLS;
    unsigned i = bi - b * NCLS;

    const float* ep = E + ((size_t)b * NN + (size_t)i * NCLS) * C + c;
    float sg[NCLS];
    float mx = -1e30f;
#pragma unroll
    for (int j = 0; j < NCLS; ++j) {
        float x = ep[j * C];
        float s = 1.f / (1.f + __expf(-x));
        sg[j] = s;
        mx = fmaxf(mx, s);
    }
    float den = 0.f;
#pragma unroll
    for (int j = 0; j < NCLS; ++j) { sg[j] = __expf(sg[j] - mx); den += sg[j]; }
    const float* up = Ujx + (size_t)b * NCLS * C + c;
    float acc = 0.f;
#pragma unroll
    for (int j = 0; j < NCLS; ++j) acc += sg[j] * up[j * C];
    Agg[idx] = acc / (den * (float)NCLS);
}

// ---------------- node update: xout = relu(xin + bn(Ux + agg)), fused bf16 copy ----------------
__global__ void node_update_kernel(const float* __restrict__ Xin,
                                   const float* __restrict__ Ux, const float* __restrict__ Agg,
                                   const float* __restrict__ Gv, const float* __restrict__ Bv,
                                   const float* __restrict__ MuV, const float* __restrict__ RsV,
                                   float* __restrict__ Xout, __bf16* __restrict__ Xbf) {
    unsigned idx = blockIdx.x * 256u + threadIdx.x;    // 0 .. NODE_ROWS*C-1
    unsigned row = idx >> 8;
    unsigned i = row % NCLS;
    float h = Ux[idx] + Agg[idx];
    float bn = Gv[i] * (h - MuV[i]) * RsV[i] + Bv[i];
    float v = fmaxf(Xin[idx] + bn, 0.f);
    Xout[idx] = v;
    Xbf[idx]  = f2bf(v);
}

// ---------------- host side ----------------
struct Ws {
    __bf16* Wbf;      // 10 * 65536 bf16
    __bf16* Xbf;      // NODE_ROWS * C
    __bf16* Ebf;      // EDGE_ROWS * C
    float *Vix, *Vjx, *Ujx, *Ux, *Agg;
    __bf16* Mbf;      // EDGE_ROWS * C
    float *muE, *rsE, *muV, *rsV;
};

static void run_layer(hipStream_t stream, const Ws& w,
                      const float* x_src, const float* edge_src,
                      const __bf16* WAb, const __bf16* WBb, const __bf16* WEb,
                      const __bf16* WUb, const __bf16* WVb,
                      const float* ge, const float* be,
                      const float* gv, const float* bv,
                      float* x_dst, float* edge_dst) {
    gemm_node4_kernel<<<dim3(NODE_ROWS / 32, 2), dim3(32, 8), 0, stream>>>(
        w.Xbf, WAb, WBb, WVb, WUb, w.Vix, w.Vjx, w.Ujx, w.Ux);
    gemm_edge_kernel<<<dim3(EDGE_ROWS / 64, 2), dim3(32, 8), 0, stream>>>(
        w.Ebf, WEb, w.Vix, w.Vjx, w.Mbf);
    stats_e_kernel<<<NN, 256, 0, stream>>>(w.Mbf, w.muE, w.rsE);
    edge_update_kernel<<<EDGE_ROWS, 256, 0, stream>>>(
        edge_src, w.Mbf, ge, be, w.muE, w.rsE, edge_dst, w.Ebf);
    agg_kernel<<<NODE_ROWS, 256, 0, stream>>>(edge_dst, w.Ujx, w.Agg);
    stats_v_kernel<<<NCLS, 256, 0, stream>>>(w.Ux, w.Agg, w.muV, w.rsV);
    node_update_kernel<<<NODE_ROWS, 256, 0, stream>>>(
        x_src, w.Ux, w.Agg, gv, bv, w.muV, w.rsV, x_dst, w.Xbf);
}

extern "C" void kernel_launch(void* const* d_in, const int* in_sizes, int n_in,
                              void* d_out, int out_size, void* d_ws, size_t ws_size,
                              hipStream_t stream) {
    (void)in_sizes; (void)n_in; (void)out_size; (void)ws_size;

    const float* x_in    = (const float*)d_in[0];
    const float* edge_in = (const float*)d_in[1];
    const float* Wsrc[10];
    for (int k = 0; k < 10; ++k) Wsrc[k] = (const float*)d_in[2 + k];   // WA1,WB1,WE1,WU1,WV1,WA2,WB2,WE2,WU2,WV2
    const float* gv1 = (const float*)d_in[12];
    const float* bv1 = (const float*)d_in[13];
    const float* ge1 = (const float*)d_in[14];
    const float* be1 = (const float*)d_in[15];
    const float* gv2 = (const float*)d_in[16];
    const float* bv2 = (const float*)d_in[17];
    const float* ge2 = (const float*)d_in[18];
    const float* be2 = (const float*)d_in[19];

    // carve workspace (256B aligned slices)
    char* p = (char*)d_ws;
    auto carve = [&](size_t bytes) -> void* {
        void* r = (void*)p;
        p += (bytes + 255) & ~(size_t)255;
        return r;
    };
    Ws w;
    w.Wbf = (__bf16*)carve((size_t)10 * C * C * sizeof(__bf16));
    w.Xbf = (__bf16*)carve((size_t)NODE_ROWS * C * sizeof(__bf16));
    w.Ebf = (__bf16*)carve((size_t)EDGE_ROWS * C * sizeof(__bf16));
    w.Vix = (float*)carve((size_t)NODE_ROWS * C * sizeof(float));
    w.Vjx = (float*)carve((size_t)NODE_ROWS * C * sizeof(float));
    w.Ujx = (float*)carve((size_t)NODE_ROWS * C * sizeof(float));
    w.Ux  = (float*)carve((size_t)NODE_ROWS * C * sizeof(float));
    w.Agg = (float*)carve((size_t)NODE_ROWS * C * sizeof(float));
    w.Mbf = (__bf16*)carve((size_t)EDGE_ROWS * C * sizeof(__bf16));
    w.muE = (float*)carve(NN * sizeof(float));
    w.rsE = (float*)carve(NN * sizeof(float));
    w.muV = (float*)carve(NCLS * sizeof(float));
    w.rsV = (float*)carve(NCLS * sizeof(float));

    // convert weights + layer-1 activations fp32 -> bf16
    for (int k = 0; k < 10; ++k)
        convert_bf16_kernel<<<(C * C / 4 + 255) / 256, 256, 0, stream>>>(
            Wsrc[k], w.Wbf + (size_t)k * C * C, C * C / 4);
    convert_bf16_kernel<<<(NODE_ROWS * C / 4 + 255) / 256, 256, 0, stream>>>(
        x_in, w.Xbf, NODE_ROWS * C / 4);
    convert_bf16_kernel<<<(EDGE_ROWS * C / 4 + 255) / 256, 256, 0, stream>>>(
        edge_in, w.Ebf, EDGE_ROWS * C / 4);

    float* x_out    = (float*)d_out;                         // [B, N, C]
    float* edge_out = (float*)d_out + (size_t)NODE_ROWS * C; // [B, N*N, C]

    const __bf16* Wb = w.Wbf;
    // layer 1: d_in -> d_out (updates also refresh Xbf/Ebf for layer 2)
    run_layer(stream, w, x_in, edge_in,
              Wb + 0 * C * C, Wb + 1 * C * C, Wb + 2 * C * C, Wb + 3 * C * C, Wb + 4 * C * C,
              ge1, be1, gv1, bv1, x_out, edge_out);
    // layer 2: in place on d_out (elementwise updates read/write the same index)
    run_layer(stream, w, x_out, edge_out,
              Wb + 5 * C * C, Wb + 6 * C * C, Wb + 7 * C * C, Wb + 8 * C * C, Wb + 9 * C * C,
              ge2, be2, gv2, bv2, x_out, edge_out);
}
